// ExpertTransformer_51522427683073
// MI455X (gfx1250) — compile-verified, matrix operand order checked by reading, not run on hardware
//
#include <hip/hip_runtime.h>
#include <hip/hip_bf16.h>
#include <math.h>

// ---------------------------------------------------------------------------
// Model constants (from the reference)
// ---------------------------------------------------------------------------
#define C_H      1024
#define C_HEADS  8
#define C_DH     128
#define C_HALF   64
#define C_L      6
#define C_ENC    2048
#define C_VOCAB  20000
#define C_KVOCAB 768
#define C_B      8
#define C_T      256
#define C_KW     64
#define C_NVIS   196

typedef __attribute__((ext_vector_type(16))) __bf16        v16bf;
typedef __attribute__((ext_vector_type(8)))  float          v8f;
typedef __attribute__((ext_vector_type(4)))  float          f32x4;
typedef __attribute__((ext_vector_type(4)))  unsigned int   u32x4;
typedef __attribute__((ext_vector_type(4)))  unsigned short u16x4;
typedef __attribute__((ext_vector_type(4)))  unsigned int   tdm_u32x4;
typedef __attribute__((ext_vector_type(8)))  int            tdm_i32x8;
typedef __attribute__((ext_vector_type(4)))  int            tdm_i32x4;

union FragBF { v16bf v; u32x4 q[2]; };

__device__ __forceinline__ unsigned short f32_to_bf16_bits(float f) {
  union { float f; unsigned int u; } c; c.f = f;
  unsigned int u = c.u;
  unsigned int r = u + 0x7FFFu + ((u >> 16) & 1u);   // round-to-nearest-even
  return (unsigned short)(r >> 16);
}

__device__ __forceinline__ float gelu_exact(float x) {
  return 0.5f * x * (1.0f + erff(x * 0.70710678118654752f));
}

// ---------------------------------------------------------------------------
// TDM: DMA one [rows<=128] x [32 bf16] tile (row-major, stride in elements)
// into LDS with 16B padding after each 64B row (=> LDS row stride 80B = LDT).
// tensor_dim0/1 are the REMAINING elements/rows from the tile start, so the
// TDM's OOB handling zero-fills every edge tile for us (no guarded path).
// ---------------------------------------------------------------------------
__device__ __forceinline__ void tdm_load_tile(unsigned lds_addr,
                                              const unsigned short* gptr,
                                              int remK, int remRows,
                                              long strideElems)
{
  unsigned long long ga = (unsigned long long)gptr;
  tdm_u32x4 g0;
  g0[0] = 1u;                                             // count=1, user mode
  g0[1] = lds_addr;                                       // LDS byte address
  g0[2] = (unsigned)(ga & 0xFFFFFFFFu);                   // global_addr[31:0]
  g0[3] = (unsigned)((ga >> 32) & 0x1FFFFFFu) | (2u << 30); // [56:32] | type=2

  unsigned td0 = (unsigned)remK;
  unsigned td1 = (unsigned)remRows;
  unsigned long long st = (unsigned long long)strideElems;
  tdm_i32x8 g1;
  g1[0] = (int)((1u << 16)        // data_size = 2 bytes
              | (1u << 20)        // pad_enable
              | (3u << 22)        // pad_interval: 16 DWORDs = 64B
              | (3u << 25));      // pad_amount: 4 DWORDs = 16B
  g1[1] = (int)((td0 & 0xFFFFu) << 16);                   // tensor_dim0[15:0]
  g1[2] = (int)((td0 >> 16) | ((td1 & 0xFFFFu) << 16));   // td0 hi | td1 lo
  g1[3] = (int)((td1 >> 16) | (32u << 16));               // td1 hi | tile_dim0
  g1[4] = (int)128;                                       // tile_dim1 = 128
  g1[5] = (int)(st & 0xFFFFFFFFu);                        // dim0 stride lo
  g1[6] = (int)((st >> 32) & 0xFFFFu);                    // dim0 stride hi
  g1[7] = 0;
  tdm_i32x4 gz = {0, 0, 0, 0};
#if defined(__clang_major__) && (__clang_major__ >= 23)
  tdm_i32x8 gz8 = {0, 0, 0, 0, 0, 0, 0, 0};
  __builtin_amdgcn_tensor_load_to_lds(g0, g1, gz, gz, gz8, 0);
#else
  __builtin_amdgcn_tensor_load_to_lds(g0, g1, gz, gz, 0);
#endif
}

// ---------------------------------------------------------------------------
// Batched WMMA GEMM:  C = alpha * A . B^T
//   A: bf16 [M,Kd] row-major (lda), B: bf16 [N,Kd] row-major (ldb).
//   OUT: 0 = f32 C[m,n] ; 1 = bf16 C[m,n] ; 2 = bf16 C[n,m] (transposed).
//   blockIdx.z decomposed as b = z/HB, h = z%HB with independent strides.
//   Tiles are DMA'd into double-buffered LDS by the Tensor Data Mover
//   (wave0 -> A tile, wave1 -> B tile), synchronized with TENSORcnt.
// ---------------------------------------------------------------------------
template<bool GELU, int OUT>
__global__ __launch_bounds__(256)
void wmma_gemm_kernel(const unsigned short* __restrict__ A, int lda, long sAb, long sAh,
                      const unsigned short* __restrict__ Bm, int ldb, long sBb, long sBh,
                      void* __restrict__ Cv, int ldc, long sCb, long sCh,
                      int M, int N, int Kd, float alpha, int HB)
{
  constexpr int BM = 128, BN = 128, BK = 32, LDT = BK + 8;  // 80B LDS row
  __shared__ unsigned short As[2][BM * LDT];
  __shared__ unsigned short Bs[2][BM * LDT];

  const int z  = blockIdx.z;
  const int bb = z / HB, hh = z % HB;
  const unsigned short* Ab  = A  + (long)bb * sAb + (long)hh * sAh;
  const unsigned short* Bbp = Bm + (long)bb * sBb + (long)hh * sBh;
  float*          Cf = (float*)Cv;
  unsigned short* Ch = (unsigned short*)Cv;
  const long cOff = (long)bb * sCb + (long)hh * sCh;

  const int m0   = blockIdx.y * BM, n0 = blockIdx.x * BN;
  const int tid  = threadIdx.x;
  const int wave = tid >> 5, lane = tid & 31;
  const int wm   = wave & 3, wn = wave >> 2;       // 4x2 wave grid
  const int mr   = lane & 15, hi = lane >> 4;      // ISA 16x32 bf16 layout

  const unsigned aBase0 = (unsigned)(unsigned long long)(void*)&As[0][0];
  const unsigned aBase1 = (unsigned)(unsigned long long)(void*)&As[1][0];
  const unsigned bBase0 = (unsigned)(unsigned long long)(void*)&Bs[0][0];
  const unsigned bBase1 = (unsigned)(unsigned long long)(void*)&Bs[1][0];

  v8f zero = {};
  v8f acc[2][4];
#pragma unroll
  for (int i = 0; i < 2; ++i)
#pragma unroll
    for (int j = 0; j < 4; ++j) acc[i][j] = zero;

  const int nTiles = (Kd + BK - 1) / BK;

  // prologue: DMA tile 0 into buffer 0
  if (wave == 0)      tdm_load_tile(aBase0, Ab + (long)m0 * lda, Kd,     M - m0, lda);
  else if (wave == 1) tdm_load_tile(bBase0, Bbp + (long)n0 * ldb, Kd,    N - n0, ldb);
  if (wave < 2) __builtin_amdgcn_s_wait_tensorcnt((short)0);
  __syncthreads();

  for (int t = 0; t < nTiles; ++t) {
    const int buf = t & 1;
    const int k1  = (t + 1) * BK;
    // issue next tile's DMA into the other buffer while computing this one
    if (k1 < Kd) {
      if (wave == 0)
        tdm_load_tile(buf ? aBase0 : aBase1, Ab + (long)m0 * lda + k1, Kd - k1, M - m0, lda);
      else if (wave == 1)
        tdm_load_tile(buf ? bBase0 : bBase1, Bbp + (long)n0 * ldb + k1, Kd - k1, N - n0, ldb);
    }

    // fragments per ISA 7.12.2: two contiguous 8-elem K chunks per lane
    FragBF af[2], bfr[4];
#pragma unroll
    for (int i = 0; i < 2; ++i) {
      const unsigned short* p = &As[buf][(wm * 32 + i * 16 + mr) * LDT + hi * 8];
      af[i].q[0] = *(const u32x4*)p;
      af[i].q[1] = *(const u32x4*)(p + 16);
    }
#pragma unroll
    for (int j = 0; j < 4; ++j) {
      const unsigned short* p = &Bs[buf][(wn * 64 + j * 16 + mr) * LDT + hi * 8];
      bfr[j].q[0] = *(const u32x4*)p;
      bfr[j].q[1] = *(const u32x4*)(p + 16);
    }
#pragma unroll
    for (int i = 0; i < 2; ++i)
#pragma unroll
      for (int j = 0; j < 4; ++j)
        acc[i][j] = __builtin_amdgcn_wmma_f32_16x16x32_bf16(
            false, af[i].v, false, bfr[j].v, (short)0, acc[i][j], false, false);

    if (k1 < Kd) {
      if (wave < 2) __builtin_amdgcn_s_wait_tensorcnt((short)0);
      __syncthreads();
    }
  }

  // epilogue: D layout = lane n (mr), VGPR r -> m = hi*8 + r
  const int mBase = m0 + wm * 32, nBase = n0 + wn * 64;
#pragma unroll
  for (int i = 0; i < 2; ++i)
#pragma unroll
    for (int j = 0; j < 4; ++j) {
      int n = nBase + j * 16 + mr;
      if (n < N) {
#pragma unroll
        for (int r = 0; r < 8; ++r) {
          int m = mBase + i * 16 + hi * 8 + r;
          if (m < M) {
            float x = alpha * acc[i][j][r];
            if (GELU) x = gelu_exact(x);
            if (OUT == 0)      Cf[cOff + (long)m * ldc + n] = x;
            else if (OUT == 1) Ch[cOff + (long)m * ldc + n] = f32_to_bf16_bits(x);
            else               Ch[cOff + (long)n * ldc + m] = f32_to_bf16_bits(x);
          }
        }
      }
    }
}

// ---------------------------------------------------------------------------
// f32 -> bf16 bulk conversion
// ---------------------------------------------------------------------------
__global__ void cvt_bf16_kernel(const float* __restrict__ in,
                                unsigned short* __restrict__ out, long n)
{
  long i = ((long)blockIdx.x * 256 + threadIdx.x) * 4;
  if (i + 3 < n) {
    f32x4 f = *(const f32x4*)(in + i);
    u16x4 h;
    h[0] = f32_to_bf16_bits(f[0]); h[1] = f32_to_bf16_bits(f[1]);
    h[2] = f32_to_bf16_bits(f[2]); h[3] = f32_to_bf16_bits(f[3]);
    *(u16x4*)(out + i) = h;
  } else {
    for (int e = 0; e < 4; ++e)
      if (i + e < n) out[i + e] = f32_to_bf16_bits(in[i + e]);
  }
}

// ---------------------------------------------------------------------------
// LayerNorm: one 256-thread block per row; bf16 output (feeds GEMMs)
// ---------------------------------------------------------------------------
__global__ __launch_bounds__(256)
void ln_kernel(const float* __restrict__ x, const float* __restrict__ g,
               const float* __restrict__ b, unsigned short* __restrict__ y, int D)
{
  __shared__ float s1[256], s2[256];
  const long row = blockIdx.x;
  const float* xr = x + row * (long)D;
  unsigned short* yr = y + row * (long)D;
  const int tid = threadIdx.x;
  float a = 0.f, q = 0.f;
  for (int i = tid; i < D; i += 256) { float v = xr[i]; a += v; q += v * v; }
  s1[tid] = a; s2[tid] = q; __syncthreads();
  for (int o = 128; o > 0; o >>= 1) {
    if (tid < o) { s1[tid] += s1[tid + o]; s2[tid] += s2[tid + o]; }
    __syncthreads();
  }
  const float mean = s1[0] / D;
  const float var  = s2[0] / D - mean * mean;
  const float rstd = rsqrtf(var + 1e-5f);
  for (int i = tid; i < D; i += 256)
    yr[i] = f32_to_bf16_bits((xr[i] - mean) * rstd * g[i] + b[i]);
}

// ---------------------------------------------------------------------------
// Embedding gather (+ optional positional embedding), f32 out
// ---------------------------------------------------------------------------
__global__ void embed_kernel(const int* __restrict__ tok, const float* __restrict__ We,
                             const float* __restrict__ wpe, float* __restrict__ out,
                             long rows, int Tmod)
{
  long i = (long)blockIdx.x * 256 + threadIdx.x;
  if (i >= rows * C_H) return;
  long row = i >> 10; int d = (int)(i & 1023);
  int t = tok[row];
  float v = We[(long)t * C_H + d];
  if (wpe) v += wpe[(long)(row % Tmod) * C_H + d];
  out[i] = v;
}

// ---------------------------------------------------------------------------
// Row softmax over f32 scores (optionally causal: valid iff n <= t)
// ---------------------------------------------------------------------------
__global__ __launch_bounds__(128)
void softmax_kernel(float* __restrict__ S, int NN, int Tmod, int causal, float scale)
{
  __shared__ float red[128];
  const long row = blockIdx.x;
  float* p = S + row * (long)NN;
  const int t = (int)(row % (long)Tmod);
  const int tid = threadIdx.x;
  float mx = -3.0e38f;
  for (int n = tid; n < NN; n += 128)
    if (!causal || n <= t) mx = fmaxf(mx, p[n] * scale);
  red[tid] = mx; __syncthreads();
  for (int o = 64; o > 0; o >>= 1) { if (tid < o) red[tid] = fmaxf(red[tid], red[tid + o]); __syncthreads(); }
  mx = red[0]; __syncthreads();
  float sum = 0.f;
  for (int n = tid; n < NN; n += 128) {
    float e = (!causal || n <= t) ? __expf(p[n] * scale - mx) : 0.f;
    p[n] = e; sum += e;
  }
  red[tid] = sum; __syncthreads();
  for (int o = 64; o > 0; o >>= 1) { if (tid < o) red[tid] += red[tid + o]; __syncthreads(); }
  const float inv = 1.f / red[0];
  for (int n = tid; n < NN; n += 128) p[n] *= inv;
}

// ---------------------------------------------------------------------------
// Differential combine: P[b,h] = S[b,2h] - lambda * S[b,2h+1], bf16 out with
// row padding NNpad (so P can be a TDM-aligned GEMM A operand).
// ---------------------------------------------------------------------------
__global__ void diffcomb_kernel(const float* __restrict__ S, unsigned short* __restrict__ P,
                                const float* __restrict__ lam, long total,
                                int Trows, int NN, int NNpad)
{
  long i = (long)blockIdx.x * 256 + threadIdx.x;
  if (i >= total) return;
  long rph = (long)Trows * NN;
  long bh = i / rph, rem = i % rph;
  long t = rem / NN, n = rem % NN;
  long b = bh >> 3, h = bh & 7;
  const float* s0 = S + (b * 16 + 2 * h)     * rph + rem;
  const float* s1 = S + (b * 16 + 2 * h + 1) * rph + rem;
  P[(bh * Trows + t) * NNpad + n] = f32_to_bf16_bits(*s0 - lam[0] * (*s1));
}

// lambda_full = exp(sum lam0*lam2) - exp(sum lam1*lam3) + 0.8  (lam: [4,64])
__global__ void lam_kernel(const float* __restrict__ lam, float* __restrict__ out)
{
  int l = threadIdx.x;  // 32 lanes
  float a = lam[l]      * lam[128 + l] + lam[32 + l] * lam[160 + l];
  float b = lam[64 + l] * lam[192 + l] + lam[96 + l] * lam[224 + l];
  for (int o = 16; o > 0; o >>= 1) { a += __shfl_xor(a, o); b += __shfl_xor(b, o); }
  if (l == 0) out[0] = __expf(a) - __expf(b) + 0.8f;
}

__global__ void add_kernel(float* __restrict__ dst, const float* __restrict__ a, long n)
{
  long i = (long)blockIdx.x * 256 + threadIdx.x;
  if (i < n) dst[i] += a[i];
}

// cls_logits[b,j] = mean_i full[b,i,j]  (i over NVIS)
__global__ void clsmean_kernel(const float* __restrict__ full, float* __restrict__ out)
{
  int i = blockIdx.x * 256 + threadIdx.x;
  if (i >= C_B * C_KVOCAB) return;
  int b = i / C_KVOCAB, j = i % C_KVOCAB;
  float s = 0.f;
  for (int v = 0; v < C_NVIS; ++v)
    s += full[((long)b * C_NVIS + v) * C_KVOCAB + j];
  out[i] = s * (1.0f / C_NVIS);
}

// probs[b] = mean_j sigmoid(cls[b,j])
__global__ __launch_bounds__(256)
void probs_kernel(const float* __restrict__ cls, float* __restrict__ out)
{
  __shared__ float red[256];
  int b = blockIdx.x, tid = threadIdx.x;
  float s = 0.f;
  for (int j = tid; j < C_KVOCAB; j += 256) {
    float x = cls[(long)b * C_KVOCAB + j];
    s += 1.0f / (1.0f + __expf(-x));
  }
  red[tid] = s; __syncthreads();
  for (int o = 128; o > 0; o >>= 1) { if (tid < o) red[tid] += red[tid + o]; __syncthreads(); }
  if (tid == 0) out[b] = red[0] * (1.0f / C_KVOCAB);
}

// ---------------------------------------------------------------------------
// Host orchestration
// ---------------------------------------------------------------------------
extern "C" void kernel_launch(void* const* d_in, const int* in_sizes, int n_in,
                              void* d_out, int out_size, void* d_ws, size_t ws_size,
                              hipStream_t stream)
{
  (void)in_sizes; (void)n_in; (void)out_size; (void)ws_size;
  // inputs in setup_inputs() dict (insertion) order, params flattened recursively
  const float* vf      = (const float*)d_in[0];
  const int*   tokens  = (const int*)  d_in[1];
  const int*   kwtok   = (const int*)  d_in[2];
  const float* We      = (const float*)d_in[3];
  const float* wpe     = (const float*)d_in[4];
  const float* clsfc   = (const float*)d_in[5];
  const float* clsproj = (const float*)d_in[6];
  const float* f_qw    = (const float*)d_in[7];
  const float* f_kw    = (const float*)d_in[8];
  const float* f_vw    = (const float*)d_in[9];
  const float* f_lam   = (const float*)d_in[10];
  const float* f_leg   = (const float*)d_in[11];
  const float* f_leb   = (const float*)d_in[12];
  const float* f_ldg   = (const float*)d_in[13];
  const float* f_ldb   = (const float*)d_in[14];
  const float* f_l2g   = (const float*)d_in[15];
  const float* f_l2b   = (const float*)d_in[16];
  const float* f_fc    = (const float*)d_in[17];
  const float* f_pj    = (const float*)d_in[18];
  const float* qw1     = (const float*)d_in[19];
  const float* kw1     = (const float*)d_in[20];
  const float* vw1     = (const float*)d_in[21];
  const float* lam1    = (const float*)d_in[22];
  const float* qw2     = (const float*)d_in[23];
  const float* kw2     = (const float*)d_in[24];
  const float* vw2     = (const float*)d_in[25];
  const float* lam2    = (const float*)d_in[26];
  const float* ln1g    = (const float*)d_in[27];
  const float* ln1b    = (const float*)d_in[28];
  const float* lneg    = (const float*)d_in[29];
  const float* lneb    = (const float*)d_in[30];
  const float* lndg    = (const float*)d_in[31];
  const float* lndb    = (const float*)d_in[32];
  const float* ln3g    = (const float*)d_in[33];
  const float* ln3b    = (const float*)d_in[34];
  const float* mfc     = (const float*)d_in[35];
  const float* mpj     = (const float*)d_in[36];

  float* outF   = (float*)d_out;
  float* logits = outF;
  float* probs  = outF + (long)C_B * C_T * C_VOCAB;          // 40,960,000
  float* clsout = probs + C_B;                               // 40,960,008

  // deterministic workspace carve (byte allocator, 256B aligned)
  char* wsp = (char*)d_ws;
  auto alloc = [&](size_t bytes) { char* p = wsp; wsp += (bytes + 255) & ~(size_t)255; return p; };
  auto cvt = [&](const float* in, unsigned short* out, long n) {
    cvt_bf16_kernel<<<(unsigned)((n / 4 + 255) / 256), 256, 0, stream>>>(in, out, n);
  };

  // bf16 weight mirrors
  unsigned short* We_bf   = (unsigned short*)alloc(2L * C_VOCAB * C_H);
  unsigned short* clsfc_b = (unsigned short*)alloc(2L * C_ENC * C_ENC);
  unsigned short* clspj_b = (unsigned short*)alloc(2L * C_KVOCAB * C_ENC);
  unsigned short* fqw_b   = (unsigned short*)alloc(2L * C_H * C_H);
  unsigned short* fkw_b   = (unsigned short*)alloc(2L * C_H * C_ENC);
  unsigned short* fvw_b   = (unsigned short*)alloc(2L * C_H * C_ENC);
  unsigned short* ffc_b   = (unsigned short*)alloc(2L * 4 * C_H * C_H);
  unsigned short* fpj_b   = (unsigned short*)alloc(2L * 4 * C_H * C_H);
  unsigned short* qw1_b   = (unsigned short*)alloc(2L * C_L * C_H * C_H);
  unsigned short* kw1_b   = (unsigned short*)alloc(2L * C_L * C_H * C_H);
  unsigned short* vw1_b   = (unsigned short*)alloc(2L * C_L * C_H * C_H);
  unsigned short* qw2_b   = (unsigned short*)alloc(2L * C_L * C_H * C_H);
  unsigned short* kw2_b   = (unsigned short*)alloc(2L * C_L * C_H * C_H);
  unsigned short* vw2_b   = (unsigned short*)alloc(2L * C_L * C_H * C_H);
  unsigned short* mfc_b   = (unsigned short*)alloc(2L * C_L * 4 * C_H * C_H);
  unsigned short* mpj_b   = (unsigned short*)alloc(2L * C_L * 4 * C_H * C_H);
  // activations
  unsigned short* vf_bf   = (unsigned short*)alloc(2L * C_B * C_NVIS * C_ENC);
  unsigned short* clsh1_b = (unsigned short*)alloc(2L * C_B * C_NVIS * C_ENC);
  float*          clsfull = (float*)         alloc(4L * C_B * C_NVIS * C_KVOCAB);
  float*          xbuf    = (float*)         alloc(4L * C_B * C_KW * C_H);
  unsigned short* xln_b   = (unsigned short*)alloc(2L * C_B * C_KW * C_H);
  unsigned short* vfln_b  = (unsigned short*)alloc(2L * C_B * C_NVIS * C_ENC);
  unsigned short* qf_b    = (unsigned short*)alloc(2L * C_B * C_KW * C_H);
  unsigned short* kf_b    = (unsigned short*)alloc(2L * C_B * C_NVIS * C_H);
  unsigned short* vtf_b   = (unsigned short*)alloc(2L * C_B * C_H * 224);   // Vt fuser, n_kv padded 196->224
  float*          Sbuf    = (float*)         alloc(4L * C_B * 16 * C_T * C_T);
  unsigned short* P_b     = (unsigned short*)alloc(2L * C_B * 8 * C_T * C_T);
  float*          tmpx    = (float*)         alloc(4L * C_B * C_KW * C_H);
  unsigned short* mlphf_b = (unsigned short*)alloc(2L * C_B * C_KW * 4 * C_H);
  float*          ybuf    = (float*)         alloc(4L * C_B * C_T * C_H);
  unsigned short* buf1_b  = (unsigned short*)alloc(2L * C_B * C_T * C_H);
  unsigned short* encln_b = (unsigned short*)alloc(2L * C_B * C_KW * C_H);
  unsigned short* q_b     = (unsigned short*)alloc(2L * C_B * C_T * C_H);
  unsigned short* k_b     = (unsigned short*)alloc(2L * C_B * C_T * C_H);
  unsigned short* vt_b    = (unsigned short*)alloc(2L * C_B * C_H * C_T);   // Vt decoder
  unsigned short* mlph_b  = (unsigned short*)alloc(2L * C_B * C_T * 4 * C_H);
  float*          buf2    = (float*)         alloc(4L * C_B * C_T * C_H);
  float*          lam     = (float*)         alloc(256);

  auto gemm = [&](int outMode, bool gelu,
                  const unsigned short* A, int lda, long sAb, long sAh,
                  const unsigned short* Bm, int ldb, long sBb, long sBh,
                  void* C, int ldc, long sCb, long sCh,
                  int M, int N, int K, float alpha, int batches, int HB) {
    dim3 g((N + 127) / 128, (M + 127) / 128, batches), blk(256);
    if (outMode == 0)
      wmma_gemm_kernel<false, 0><<<g, blk, 0, stream>>>(A, lda, sAb, sAh, Bm, ldb, sBb, sBh, C, ldc, sCb, sCh, M, N, K, alpha, HB);
    else if (outMode == 1 && gelu)
      wmma_gemm_kernel<true,  1><<<g, blk, 0, stream>>>(A, lda, sAb, sAh, Bm, ldb, sBb, sBh, C, ldc, sCb, sCh, M, N, K, alpha, HB);
    else if (outMode == 1)
      wmma_gemm_kernel<false, 1><<<g, blk, 0, stream>>>(A, lda, sAb, sAh, Bm, ldb, sBb, sBh, C, ldc, sCb, sCh, M, N, K, alpha, HB);
    else
      wmma_gemm_kernel<false, 2><<<g, blk, 0, stream>>>(A, lda, sAb, sAh, Bm, ldb, sBb, sBh, C, ldc, sCb, sCh, M, N, K, alpha, HB);
  };
  auto add = [&](float* dst, const float* a, long n) {
    add_kernel<<<(unsigned)((n + 255) / 256), 256, 0, stream>>>(dst, a, n);
  };
  auto ln = [&](const float* x, const float* g, const float* b, unsigned short* y, int rows, int D) {
    ln_kernel<<<rows, 256, 0, stream>>>(x, g, b, y, D);
  };

  // ---------------- weight / input conversion to bf16 ----------------
  cvt(We,      We_bf,   (long)C_VOCAB * C_H);
  cvt(clsfc,   clsfc_b, (long)C_ENC * C_ENC);
  cvt(clsproj, clspj_b, (long)C_KVOCAB * C_ENC);
  cvt(f_qw,    fqw_b,   (long)C_H * C_H);
  cvt(f_kw,    fkw_b,   (long)C_H * C_ENC);
  cvt(f_vw,    fvw_b,   (long)C_H * C_ENC);
  cvt(f_fc,    ffc_b,   4L * C_H * C_H);
  cvt(f_pj,    fpj_b,   4L * C_H * C_H);
  cvt(qw1,     qw1_b,   (long)C_L * C_H * C_H);
  cvt(kw1,     kw1_b,   (long)C_L * C_H * C_H);
  cvt(vw1,     vw1_b,   (long)C_L * C_H * C_H);
  cvt(qw2,     qw2_b,   (long)C_L * C_H * C_H);
  cvt(kw2,     kw2_b,   (long)C_L * C_H * C_H);
  cvt(vw2,     vw2_b,   (long)C_L * C_H * C_H);
  cvt(mfc,     mfc_b,   (long)C_L * 4 * C_H * C_H);
  cvt(mpj,     mpj_b,   (long)C_L * 4 * C_H * C_H);
  cvt(vf,      vf_bf,   (long)C_B * C_NVIS * C_ENC);

  // ---------------- Phase 1: classifier head ----------------
  gemm(1, true,  vf_bf,   C_ENC, 0, 0, clsfc_b, C_ENC, 0, 0,
       clsh1_b, C_ENC, 0, 0, C_B * C_NVIS, C_ENC, C_ENC, 1.f, 1, 1);
  gemm(0, false, clsh1_b, C_ENC, 0, 0, clspj_b, C_ENC, 0, 0,
       clsfull, C_KVOCAB, 0, 0, C_B * C_NVIS, C_KVOCAB, C_ENC, 1.f, 1, 1);
  clsmean_kernel<<<(C_B * C_KVOCAB + 255) / 256, 256, 0, stream>>>(clsfull, clsout);
  probs_kernel<<<C_B, 256, 0, stream>>>(clsout, probs);

  // ---------------- Phase 2: ImageKeywordFuser ----------------
  const long rowsKW = (long)C_B * C_KW;                       // 512
  embed_kernel<<<(unsigned)((rowsKW * C_H + 255) / 256), 256, 0, stream>>>(
      kwtok, We, nullptr, xbuf, rowsKW, 1);
  ln(xbuf, f_ldg, f_ldb, xln_b, (int)rowsKW, C_H);
  ln(vf, f_leg, f_leb, vfln_b, C_B * C_NVIS, C_ENC);
  gemm(1, false, xln_b,  C_H,   0, 0, fqw_b, C_H,   0, 0, qf_b, C_H, 0, 0, (int)rowsKW,  C_H, C_H,   1.f, 1, 1);
  gemm(1, false, vfln_b, C_ENC, 0, 0, fkw_b, C_ENC, 0, 0, kf_b, C_H, 0, 0, C_B * C_NVIS, C_H, C_ENC, 1.f, 1, 1);
  // Vt: per-image batched, transposed store -> [b][h*128+d][n_kv(pad 224)]
  gemm(2, false, vfln_b, C_ENC, (long)C_NVIS * C_ENC, 0, fvw_b, C_ENC, 0, 0,
       vtf_b, 224, (long)C_H * 224, 0, C_NVIS, C_H, C_ENC, 1.f, C_B, 1);
  lam_kernel<<<1, 32, 0, stream>>>(f_lam, lam);
  gemm(0, false,
       qf_b, C_H, (long)C_KW * C_H,   C_HALF,
       kf_b, C_H, (long)C_NVIS * C_H, C_HALF,
       Sbuf, C_NVIS, (long)16 * C_KW * C_NVIS, (long)C_KW * C_NVIS,
       C_KW, C_NVIS, C_HALF, 1.f, C_B * 16, 16);
  softmax_kernel<<<C_B * 16 * C_KW, 128, 0, stream>>>(Sbuf, C_NVIS, C_KW, 0, 0.125f);
  diffcomb_kernel<<<(unsigned)(((long)C_B * 8 * C_KW * C_NVIS + 255) / 256), 256, 0, stream>>>(
      Sbuf, P_b, lam, (long)C_B * 8 * C_KW * C_NVIS, C_KW, C_NVIS, 256);
  gemm(0, false,
       P_b,   256, (long)8 * C_KW * 256, (long)C_KW * 256,
       vtf_b, 224, (long)C_H * 224,      (long)C_DH * 224,
       tmpx,  C_H, (long)C_KW * C_H,     C_DH,
       C_KW, C_DH, C_NVIS, 1.f, C_B * 8, 8);
  add(xbuf, tmpx, rowsKW * C_H);
  ln(xbuf, f_l2g, f_l2b, xln_b, (int)rowsKW, C_H);
  gemm(1, true,  xln_b,   C_H,     0, 0, ffc_b, C_H,     0, 0, mlphf_b, 4 * C_H, 0, 0, (int)rowsKW, 4 * C_H, C_H,     1.f, 1, 1);
  gemm(0, false, mlphf_b, 4 * C_H, 0, 0, fpj_b, 4 * C_H, 0, 0, tmpx,    C_H,     0, 0, (int)rowsKW, C_H,     4 * C_H, 1.f, 1, 1);
  add(xbuf, tmpx, rowsKW * C_H);   // enc := xbuf

  // ---------------- Phase 3: decoder stack ----------------
  const long rowsBT = (long)C_B * C_T;                        // 2048
  embed_kernel<<<(unsigned)((rowsBT * C_H + 255) / 256), 256, 0, stream>>>(
      tokens, We, wpe, ybuf, rowsBT, C_T);

  for (int i = 0; i < C_L; ++i) {
    const long wOff = (long)i * C_H * C_H;
    const long mOff = (long)i * 4 * C_H * C_H;
    const long nOff = (long)i * C_H;
    const long lOff = (long)i * 4 * C_HALF;

    // --- self differential attention (replaces y) ---
    ln(ybuf, ln1g + nOff, ln1b + nOff, buf1_b, (int)rowsBT, C_H);
    gemm(1, false, buf1_b, C_H, 0, 0, qw1_b + wOff, C_H, 0, 0, q_b, C_H, 0, 0, (int)rowsBT, C_H, C_H, 1.f, 1, 1);
    gemm(1, false, buf1_b, C_H, 0, 0, kw1_b + wOff, C_H, 0, 0, k_b, C_H, 0, 0, (int)rowsBT, C_H, C_H, 1.f, 1, 1);
    gemm(2, false, buf1_b, C_H, (long)C_T * C_H, 0, vw1_b + wOff, C_H, 0, 0,
         vt_b, C_T, (long)C_H * C_T, 0, C_T, C_H, C_H, 1.f, C_B, 1);
    lam_kernel<<<1, 32, 0, stream>>>(lam1 + lOff, lam);
    gemm(0, false,
         q_b, C_H, (long)C_T * C_H, C_HALF,
         k_b, C_H, (long)C_T * C_H, C_HALF,
         Sbuf, C_T, (long)16 * C_T * C_T, (long)C_T * C_T,
         C_T, C_T, C_HALF, 1.f, C_B * 16, 16);
    softmax_kernel<<<C_B * 16 * C_T, 128, 0, stream>>>(Sbuf, C_T, C_T, 1, 0.125f);
    diffcomb_kernel<<<(unsigned)(((long)C_B * 8 * C_T * C_T + 255) / 256), 256, 0, stream>>>(
        Sbuf, P_b, lam, (long)C_B * 8 * C_T * C_T, C_T, C_T, 256);
    gemm(0, false,
         P_b,  256, (long)8 * C_T * 256, (long)C_T * 256,
         vt_b, C_T, (long)C_H * C_T,     (long)C_DH * C_T,
         ybuf, C_H, (long)C_T * C_H,     C_DH,
         C_T, C_DH, C_T, 1.f, C_B * 8, 8);

    // --- cross differential attention vs enc (replaces y) ---
    ln(ybuf, lndg + nOff, lndb + nOff, buf1_b, (int)rowsBT, C_H);
    ln(xbuf, lneg + nOff, lneb + nOff, encln_b, (int)rowsKW, C_H);
    gemm(1, false, buf1_b,  C_H, 0, 0, qw2_b + wOff, C_H, 0, 0, q_b, C_H, 0, 0, (int)rowsBT, C_H, C_H, 1.f, 1, 1);
    gemm(1, false, encln_b, C_H, 0, 0, kw2_b + wOff, C_H, 0, 0, k_b, C_H, 0, 0, (int)rowsKW, C_H, C_H, 1.f, 1, 1);
    gemm(2, false, encln_b, C_H, (long)C_KW * C_H, 0, vw2_b + wOff, C_H, 0, 0,
         vt_b, C_KW, (long)C_H * C_KW, 0, C_KW, C_H, C_H, 1.f, C_B, 1);
    lam_kernel<<<1, 32, 0, stream>>>(lam2 + lOff, lam);
    gemm(0, false,
         q_b, C_H, (long)C_T * C_H,  C_HALF,
         k_b, C_H, (long)C_KW * C_H, C_HALF,
         Sbuf, C_KW, (long)16 * C_T * C_KW, (long)C_T * C_KW,
         C_T, C_KW, C_HALF, 1.f, C_B * 16, 16);
    softmax_kernel<<<C_B * 16 * C_T, 128, 0, stream>>>(Sbuf, C_KW, C_T, 0, 0.125f);
    diffcomb_kernel<<<(unsigned)(((long)C_B * 8 * C_T * C_KW + 255) / 256), 256, 0, stream>>>(
        Sbuf, P_b, lam, (long)C_B * 8 * C_T * C_KW, C_T, C_KW, C_KW);
    gemm(0, false,
         P_b,  C_KW, (long)8 * C_T * C_KW, (long)C_T * C_KW,
         vt_b, C_KW, (long)C_H * C_KW,     (long)C_DH * C_KW,
         ybuf, C_H,  (long)C_T * C_H,      C_DH,
         C_T, C_DH, C_KW, 1.f, C_B * 8, 8);

    // --- MLP with residual ---
    ln(ybuf, ln3g + nOff, ln3b + nOff, buf1_b, (int)rowsBT, C_H);
    gemm(1, true,  buf1_b, C_H,     0, 0, mfc_b + mOff, C_H,     0, 0, mlph_b, 4 * C_H, 0, 0, (int)rowsBT, 4 * C_H, C_H,     1.f, 1, 1);
    gemm(0, false, mlph_b, 4 * C_H, 0, 0, mpj_b + mOff, 4 * C_H, 0, 0, buf2,   C_H,     0, 0, (int)rowsBT, C_H,     4 * C_H, 1.f, 1, 1);
    add(ybuf, buf2, rowsBT * C_H);
  }

  // ---------------- tied LM head ----------------
  cvt(ybuf, buf1_b, rowsBT * C_H);
  gemm(0, false, buf1_b, C_H, 0, 0, We_bf, C_H, 0, 0,
       logits, C_VOCAB, 0, 0, (int)rowsBT, C_VOCAB, C_H, 1.f, 1, 1);
}
